// RWKV_49838800503455
// MI455X (gfx1250) — matrix operand order repaired and checked by reading, not compiled, not run on hardware
//
#include <hip/hip_runtime.h>
#include <hip/hip_bf16.h>
#include <math.h>

typedef __attribute__((ext_vector_type(16))) _Float16 v16h;
typedef __attribute__((ext_vector_type(8)))  float    v8f;

#define LN_EPS 1e-5f
#define GN_EPS 0.00064f

__device__ __forceinline__ float warp_sum(float v) {
#pragma unroll
  for (int off = 16; off; off >>= 1) v += __shfl_xor(v, off, 32);
  return v;
}

// ---------------- embedding gather + LayerNorm (wave per row) ----------------
__global__ void embed_ln_kernel(const int* __restrict__ idx, const float* __restrict__ emb,
                                const float* __restrict__ w, const float* __restrict__ b,
                                float* __restrict__ out, int rows, int C) {
  int wid = (blockIdx.x * blockDim.x + threadIdx.x) >> 5;
  int lane = threadIdx.x & 31;
  if (wid >= rows) return;
  const float* src = emb + (size_t)idx[wid] * C;
  float s = 0.f, s2 = 0.f;
  for (int c = lane; c < C; c += 32) { float v = src[c]; s += v; s2 += v * v; }
  s = warp_sum(s); s2 = warp_sum(s2);
  float mean = s / C, var = s2 / C - mean * mean;
  float rstd = rsqrtf(var + LN_EPS);
  float* dst = out + (size_t)wid * C;
  for (int c = lane; c < C; c += 32) dst[c] = (src[c] - mean) * rstd * w[c] + b[c];
}

// ---------------- LayerNorm over rows (wave per row) ----------------
__global__ void ln_kernel(const float* __restrict__ in, const float* __restrict__ w,
                          const float* __restrict__ b, float* __restrict__ out, int rows, int C) {
  int wid = (blockIdx.x * blockDim.x + threadIdx.x) >> 5;
  int lane = threadIdx.x & 31;
  if (wid >= rows) return;
  const float* src = in + (size_t)wid * C;
  float s = 0.f, s2 = 0.f;
  for (int c = lane; c < C; c += 32) { float v = src[c]; s += v; s2 += v * v; }
  s = warp_sum(s); s2 = warp_sum(s2);
  float mean = s / C, var = s2 / C - mean * mean;
  float rstd = rsqrtf(var + LN_EPS);
  float* dst = out + (size_t)wid * C;
  for (int c = lane; c < C; c += 32) dst[c] = (src[c] - mean) * rstd * w[c] + b[c];
}

// ---------------- time-mix token shift: 6 lerps ----------------
__global__ void tmix_kernel(const float* __restrict__ xn,
                            const float* mr, const float* mw, const float* mk,
                            const float* mv, const float* ma, const float* mg,
                            float* xr, float* xw, float* xk,
                            float* xv, float* xa, float* xg,
                            int T, int C, int total) {
  int i = blockIdx.x * blockDim.x + threadIdx.x;
  if (i >= total) return;
  int c = i % C; int m = i / C; int t = m % T;
  float cur = xn[i];
  float prev = (t > 0) ? xn[i - C] : 0.f;
  float xx = prev - cur;
  xr[i] = cur + xx * mr[c]; xw[i] = cur + xx * mw[c]; xk[i] = cur + xx * mk[c];
  xv[i] = cur + xx * mv[c]; xa[i] = cur + xx * ma[c]; xg[i] = cur + xx * mg[c];
}

// ---------------- channel-mix token shift ----------------
__global__ void cmix_kernel(const float* __restrict__ xn, const float* __restrict__ fxk,
                            float* __restrict__ kx, int T, int C, int total) {
  int i = blockIdx.x * blockDim.x + threadIdx.x;
  if (i >= total) return;
  int c = i % C; int m = i / C; int t = m % T;
  float cur = xn[i];
  float prev = (t > 0) ? xn[i - C] : 0.f;
  kx[i] = cur + (prev - cur) * fxk[c];
}

// ---------------- LDS-staged WMMA GEMM: C = epilogue(A[M,K] @ B[K,N]) ------
// Workgroup (256 thr = 8 waves) computes a 128 x NBLK tile.
// Per K-chunk of 32: stage A(128x32, row-major) + B(32xNBLK, TRANSPOSED to
// n-major fragment-ready order) into LDS as f16. Fragments then read as
// contiguous 32-byte runs -> ds_load_b128 pairs for both A and B.
// Each wave runs NT = NBLK/16 v_wmma_f32_16x16x32_f16 sharing one A fragment.
// EPI: 0=store, 1=tanh, 2=sigmoid, 3=relu^2, 4=accumulate into C
template <int NBLK, int EPI>
__global__ void gemm_wmma_kernel(const float* __restrict__ A, const float* __restrict__ Bw,
                                 float* __restrict__ Cm, int M, int N, int K) {
  constexpr int NT    = NBLK / 16;    // wmma tiles per wave in N
  constexpr int ASTR  = 32 + 8;       // padded LDS stride (f16), 80B rows
  constexpr int BTSTR = 32 + 8;       // padded k-stride for transposed B
  constexpr int NW    = NBLK / 16;    // B floats per thread per k-row (4 or 2)
  __shared__ _Float16 Alds[128 * ASTR];
  __shared__ _Float16 BldsT[NBLK * BTSTR];   // [n][k] fragment-ready

  int tid  = threadIdx.x;
  int lane = tid & 31;
  int wave = tid >> 5;                // 0..7
  int half = lane >> 4;
  int l15  = lane & 15;

  int nTilesN = N / NBLK;
  int tM = blockIdx.x / nTilesN;
  int tN = blockIdx.x % nTilesN;
  int m0 = tM * 128;
  int n0 = tN * NBLK;

  // A staging: thread -> (row, 16-float half-row), coalesced float4 loads
  int aRow = tid >> 1;                // 0..127
  int aOff = (tid & 1) * 16;          // 0 or 16
  const float* aSrc = A + (size_t)(m0 + aRow) * K + aOff;

  // B staging: thread -> k-pair (2kp, 2kp+1) x NW consecutive n columns.
  // Consecutive lanes cover consecutive n within a k-row pair (coalesced).
  int kp   = tid >> 4;                // 0..15
  int nOff = (tid & 15) * NW;
  const float* bSrc0 = Bw + (size_t)(2 * kp) * N + n0 + nOff;
  const float* bSrc1 = bSrc0 + N;

  v8f zero = {};
  v8f acc[NT];
#pragma unroll
  for (int t = 0; t < NT; t++) acc[t] = zero;

  for (int kb = 0; kb < K; kb += 32) {
    // ---- stage A tile (row-major f16) ----
    const float4* a4 = (const float4*)(aSrc + kb);
#pragma unroll
    for (int q = 0; q < 4; q++) {
      float4 v = a4[q];
      int kk = aOff + q * 4;
      _Float16* dst = &Alds[aRow * ASTR + kk];
      dst[0] = (_Float16)v.x; dst[1] = (_Float16)v.y;
      dst[2] = (_Float16)v.z; dst[3] = (_Float16)v.w;
    }
    // ---- stage B tile transposed: BldsT[n][k], k-pairs merge to b32 stores ----
    size_t bk = (size_t)kb * N;
#pragma unroll
    for (int j = 0; j < NW; j++) {
      _Float16 h0 = (_Float16)bSrc0[bk + j];
      _Float16 h1 = (_Float16)bSrc1[bk + j];
      _Float16* d = &BldsT[(nOff + j) * BTSTR + 2 * kp];
      d[0] = h0; d[1] = h1;
    }
    // prefetch next K chunk while this one is consumed
    if (kb + 32 < K) {
      __builtin_prefetch(aSrc + kb + 32, 0, 0);
      __builtin_prefetch(bSrc0 + bk + N + N, 0, 0);
    }
    __syncthreads();

    // ---- A fragment: contiguous 16-f16 run (k = half*16..half*16+15) ----
    v16h af;
    int arow = wave * 16 + l15;
#pragma unroll
    for (int e = 0; e < 16; e++) {
      int k = (half << 4) + e;
      af[e] = Alds[arow * ASTR + k];
    }
    // ---- B fragments: 16 contiguous f16 per lane -> 2x ds_load_b128 ----
#pragma unroll
    for (int t = 0; t < NT; t++) {
      const _Float16* bp = &BldsT[(t * 16 + l15) * BTSTR + half * 16];
      v16h bf;
#pragma unroll
      for (int e = 0; e < 16; e++) bf[e] = bp[e];
      acc[t] = __builtin_amdgcn_wmma_f32_16x16x32_f16(false, af, false, bf,
                                                      (short)0, acc[t], false, false);
    }
    __syncthreads();
  }

  // ---- epilogue (C/D layout: lane=N col, vgpr r -> M = half*8 + r) ----
  int mBase = m0 + wave * 16 + half * 8;
#pragma unroll
  for (int t = 0; t < NT; t++) {
#pragma unroll
    for (int rI = 0; rI < 8; rI++) {
      float v = acc[t][rI];
      size_t ix = (size_t)(mBase + rI) * N + n0 + t * 16 + l15;
      if (EPI == 0)      Cm[ix] = v;
      else if (EPI == 1) Cm[ix] = tanhf(v);
      else if (EPI == 2) Cm[ix] = 1.f / (1.f + expf(-v));
      else if (EPI == 3) Cm[ix] = v > 0.f ? v * v : 0.f;
      else               Cm[ix] += v;
    }
  }
}

// ---------------- LoRA post-ops ----------------
__global__ void ew_w_kernel(const float* __restrict__ lt, const float* __restrict__ w0,
                            float* __restrict__ wr, int C, int total) {
  int i = blockIdx.x * blockDim.x + threadIdx.x;
  if (i >= total) return;
  int c = i % C;
  float z = w0[c] + lt[i];
  wr[i] = -log1pf(expf(-z)) - 0.5f;       // -softplus(-z) - 0.5
}
__global__ void ew_a_kernel(const float* __restrict__ lt, const float* __restrict__ a0,
                            float* __restrict__ aa, int C, int total) {
  int i = blockIdx.x * blockDim.x + threadIdx.x;
  if (i >= total) return;
  int c = i % C;
  aa[i] = 1.f / (1.f + expf(-(a0[c] + lt[i])));
}
__global__ void ew_vmix_kernel(const float* __restrict__ lt, const float* __restrict__ v0,
                               const float* __restrict__ vfirst, float* __restrict__ vb,
                               int C, int total) {
  int i = blockIdx.x * blockDim.x + threadIdx.x;
  if (i >= total) return;
  int c = i % C;
  float s = 1.f / (1.f + expf(-(v0[c] + lt[i])));
  float v = vb[i];
  vb[i] = v + (vfirst[i] - v) * s;
}
__global__ void copy_kernel(const float* __restrict__ a, float* __restrict__ b, int total) {
  int i = blockIdx.x * blockDim.x + threadIdx.x;
  if (i < total) b[i] = a[i];
}

// ---------------- kk normalize + k rescale (wave per (row,head), N=64) ----------------
__global__ void kk_prep_kernel(float* __restrict__ kbuf, const float* __restrict__ aa,
                               const float* __restrict__ k_k, const float* __restrict__ k_a,
                               float* __restrict__ mkk, float* __restrict__ kka,
                               int rows, int H, int N) {
  int wid = (blockIdx.x * blockDim.x + threadIdx.x) >> 5;
  int lane = threadIdx.x & 31;
  if (wid >= rows * H) return;
  int m = wid / H, h = wid % H;
  size_t base = (size_t)m * (H * N) + (size_t)h * N;
  int c0 = h * N;
  int n0 = lane, n1 = lane + 32;
  float k0 = kbuf[base + n0], k1 = kbuf[base + n1];
  float kk0 = k0 * k_k[c0 + n0], kk1 = k1 * k_k[c0 + n1];
  float ss = warp_sum(kk0 * kk0 + kk1 * kk1);
  float nrm = sqrtf(ss);
  nrm = nrm > 1e-12f ? nrm : 1e-12f;
  float inv = 1.f / nrm;
  float a0v = aa[base + n0], a1v = aa[base + n1];
  mkk[base + n0] = -kk0 * inv;       mkk[base + n1] = -kk1 * inv;
  kka[base + n0] = kk0 * inv * a0v;  kka[base + n1] = kk1 * inv * a1v;
  kbuf[base + n0] = k0 * (1.f + (a0v - 1.f) * k_a[c0 + n0]);
  kbuf[base + n1] = k1 * (1.f + (a1v - 1.f) * k_a[c0 + n1]);
}

// ---------------- RWKV7 sequential scan: one block per (b,h), 64 threads ----------------
__global__ void scan_kernel(const float* __restrict__ r, const float* __restrict__ wraw,
                            const float* __restrict__ k, const float* __restrict__ v,
                            const float* __restrict__ am, const float* __restrict__ bm,
                            float* __restrict__ y, int T, int H, int C) {
  int b = blockIdx.x / H, h = blockIdx.x % H;
  int i = threadIdx.x;                           // 0..63 (row of S)
  __shared__ float sr[64], sw[64], sk[64], sv[64], sa[64], sb[64];
  float S[64];
#pragma unroll
  for (int j = 0; j < 64; j++) S[j] = 0.f;
  for (int t = 0; t < T; t++) {
    size_t base = ((size_t)(b * T + t)) * C + (size_t)h * 64;
    sr[i] = r[base + i];
    sw[i] = expf(-expf(wraw[base + i]));         // decay
    sk[i] = k[base + i]; sv[i] = v[base + i];
    sa[i] = am[base + i]; sb[i] = bm[base + i];
    __syncthreads();
    float sai = 0.f;
#pragma unroll
    for (int j = 0; j < 64; j++) sai += S[j] * sa[j];
    float vi = sv[i], yi = 0.f;
#pragma unroll
    for (int j = 0; j < 64; j++) {
      float s = S[j] * sw[j] + sai * sb[j] + vi * sk[j];
      S[j] = s;
      yi += s * sr[j];
    }
    y[base + i] = yi;
    __syncthreads();
  }
}

// ---------------- GroupNorm + rk*v + gate (wave per (row,head), N=64) ----------------
__global__ void post_y_kernel(const float* __restrict__ y, const float* __restrict__ r,
                              const float* __restrict__ k, const float* __restrict__ v,
                              const float* __restrict__ gg, const float* __restrict__ gw,
                              const float* __restrict__ gb, const float* __restrict__ r_k,
                              float* __restrict__ out, int rows, int H, int N) {
  int wid = (blockIdx.x * blockDim.x + threadIdx.x) >> 5;
  int lane = threadIdx.x & 31;
  if (wid >= rows * H) return;
  int m = wid / H, h = wid % H;
  size_t base = (size_t)m * (H * N) + (size_t)h * N;
  int c0 = h * N;
  int n0 = lane, n1 = lane + 32;
  float y0 = y[base + n0], y1 = y[base + n1];
  float s  = warp_sum(y0 + y1);
  float s2 = warp_sum(y0 * y0 + y1 * y1);
  float mean = s / (float)N, var = s2 / (float)N - mean * mean;
  float rstd = rsqrtf(var + GN_EPS);
  float rk = warp_sum(r[base + n0] * k[base + n0] * r_k[c0 + n0] +
                      r[base + n1] * k[base + n1] * r_k[c0 + n1]);
  float g0 = (y0 - mean) * rstd * gw[c0 + n0] + gb[c0 + n0];
  float g1 = (y1 - mean) * rstd * gw[c0 + n1] + gb[c0 + n1];
  out[base + n0] = (g0 + rk * v[base + n0]) * gg[base + n0];
  out[base + n1] = (g1 + rk * v[base + n1]) * gg[base + n1];
}

// ---------------- final head: out[row] = dot(xn[row], head_w) (wave per row) ----------------
__global__ void head_kernel(const float* __restrict__ xn, const float* __restrict__ hw,
                            float* __restrict__ out, int rows, int C) {
  int wid = (blockIdx.x * blockDim.x + threadIdx.x) >> 5;
  int lane = threadIdx.x & 31;
  if (wid >= rows) return;
  const float* src = xn + (size_t)wid * C;
  float s = 0.f;
  for (int c = lane; c < C; c += 32) s += src[c] * hw[c];
  s = warp_sum(s);
  if (lane == 0) out[wid] = s;
}

// =====================================================================

static void launch_gemm(const float* A, const float* Bw, float* Cm,
                        int M, int N, int K, int epi, hipStream_t stream) {
  dim3 blk(256);
  if (N % 64 == 0) {
    dim3 g((M / 128) * (N / 64));
    switch (epi) {
      case 0: gemm_wmma_kernel<64, 0><<<g, blk, 0, stream>>>(A, Bw, Cm, M, N, K); break;
      case 1: gemm_wmma_kernel<64, 1><<<g, blk, 0, stream>>>(A, Bw, Cm, M, N, K); break;
      case 2: gemm_wmma_kernel<64, 2><<<g, blk, 0, stream>>>(A, Bw, Cm, M, N, K); break;
      case 3: gemm_wmma_kernel<64, 3><<<g, blk, 0, stream>>>(A, Bw, Cm, M, N, K); break;
      default:gemm_wmma_kernel<64, 4><<<g, blk, 0, stream>>>(A, Bw, Cm, M, N, K); break;
    }
  } else {
    dim3 g((M / 128) * (N / 32));
    switch (epi) {
      case 0: gemm_wmma_kernel<32, 0><<<g, blk, 0, stream>>>(A, Bw, Cm, M, N, K); break;
      case 1: gemm_wmma_kernel<32, 1><<<g, blk, 0, stream>>>(A, Bw, Cm, M, N, K); break;
      case 2: gemm_wmma_kernel<32, 2><<<g, blk, 0, stream>>>(A, Bw, Cm, M, N, K); break;
      case 3: gemm_wmma_kernel<32, 3><<<g, blk, 0, stream>>>(A, Bw, Cm, M, N, K); break;
      default:gemm_wmma_kernel<32, 4><<<g, blk, 0, stream>>>(A, Bw, Cm, M, N, K); break;
    }
  }
}

extern "C" void kernel_launch(void* const* d_in, const int* in_sizes, int n_in,
                              void* d_out, int out_size, void* d_ws, size_t ws_size,
                              hipStream_t stream) {
  (void)in_sizes; (void)n_in; (void)out_size; (void)ws_size;

  // Inputs in setup_inputs() order
  const int*   idx     = (const int*)  d_in[0];
  const float* emb     = (const float*)d_in[1];
  const float* ln0_w   = (const float*)d_in[2];
  const float* ln0_b   = (const float*)d_in[3];
  const float* ln1_w   = (const float*)d_in[4];
  const float* ln1_b   = (const float*)d_in[5];
  const float* ln2_w   = (const float*)d_in[6];
  const float* ln2_b   = (const float*)d_in[7];
  const float* x_r     = (const float*)d_in[8];
  const float* x_w     = (const float*)d_in[9];
  const float* x_k     = (const float*)d_in[10];
  const float* x_v     = (const float*)d_in[11];
  const float* x_a     = (const float*)d_in[12];
  const float* x_g     = (const float*)d_in[13];
  const float* w0      = (const float*)d_in[14];
  const float* w1      = (const float*)d_in[15];
  const float* w2      = (const float*)d_in[16];
  const float* a0      = (const float*)d_in[17];
  const float* a1      = (const float*)d_in[18];
  const float* a2      = (const float*)d_in[19];
  const float* v0      = (const float*)d_in[20];
  const float* v1      = (const float*)d_in[21];
  const float* v2      = (const float*)d_in[22];
  const float* g1      = (const float*)d_in[23];
  const float* g2      = (const float*)d_in[24];
  const float* k_k     = (const float*)d_in[25];
  const float* k_a     = (const float*)d_in[26];
  const float* r_k     = (const float*)d_in[27];
  const float* Wr      = (const float*)d_in[28];
  const float* Wk      = (const float*)d_in[29];
  const float* Wv      = (const float*)d_in[30];
  const float* Wo      = (const float*)d_in[31];
  const float* lnx_w   = (const float*)d_in[32];
  const float* lnx_b   = (const float*)d_in[33];
  const float* f_xk    = (const float*)d_in[34];
  const float* f_key   = (const float*)d_in[35];
  const float* f_val   = (const float*)d_in[36];
  const float* lnout_w = (const float*)d_in[37];
  const float* lnout_b = (const float*)d_in[38];
  const float* head_w  = (const float*)d_in[39];

  constexpr int B = 4, T = 512, C = 896, H = 14, N = 64;
  constexpr int D = 64, DG = 160, F4 = 4 * C, L = 2;
  const int M = B * T;
  const size_t MC = (size_t)M * C;
  const size_t CC = (size_t)C * C;

  // Workspace layout (fp32)
  float* p = (float*)d_ws;
  auto alloc = [&](size_t n) { float* r_ = p; p += n; return r_; };
  float* x      = alloc(MC);
  float* xn     = alloc(MC);
  float* xr     = alloc(MC);   // also reused as gated y ("yg")
  float* xw     = alloc(MC);
  float* xk     = alloc(MC);   // also reused as channel-mix kx
  float* xv     = alloc(MC);
  float* xa     = alloc(MC);
  float* xg     = alloc(MC);
  float* rb     = alloc(MC);
  float* kb     = alloc(MC);
  float* vb     = alloc(MC);
  float* yb     = alloc(MC);
  float* wrw    = alloc(MC);
  float* aa     = alloc(MC);
  float* gg     = alloc(MC);
  float* mkk    = alloc(MC);
  float* kka    = alloc(MC);
  float* vfirst = alloc(MC);
  float* lt     = alloc(MC);
  float* t1     = alloc((size_t)M * DG);
  float* hbuf   = alloc((size_t)M * F4);

  const int rowGrid  = (M + 7) / 8;                 // wave-per-row kernels, 256 thr = 8 waves
  const int mhGrid   = (M * H + 7) / 8;             // wave-per-(row,head)
  const int ewGrid   = (int)((MC + 255) / 256);     // elementwise over M*C
  dim3 blk256(256);

  // x = LN0(emb[idx])
  embed_ln_kernel<<<rowGrid, blk256, 0, stream>>>(idx, emb, ln0_w, ln0_b, x, M, C);

  for (int l = 0; l < L; l++) {
    const size_t lC = (size_t)l * C;
    // ---- time mix ----
    ln_kernel<<<rowGrid, blk256, 0, stream>>>(x, ln1_w + lC, ln1_b + lC, xn, M, C);
    tmix_kernel<<<ewGrid, blk256, 0, stream>>>(xn,
        x_r + lC, x_w + lC, x_k + lC, x_v + lC, x_a + lC, x_g + lC,
        xr, xw, xk, xv, xa, xg, T, C, (int)MC);

    launch_gemm(xr, Wr + (size_t)l * CC, rb, M, C, C, 0, stream);
    launch_gemm(xk, Wk + (size_t)l * CC, kb, M, C, C, 0, stream);
    launch_gemm(xv, Wv + (size_t)l * CC, vb, M, C, C, 0, stream);

    // w LoRA: wraw = -softplus(-(w0 + tanh(xw@w1)@w2)) - 0.5
    launch_gemm(xw, w1 + (size_t)l * C * D, t1, M, D, C, 1, stream);
    launch_gemm(t1, w2 + (size_t)l * D * C, lt, M, C, D, 0, stream);
    ew_w_kernel<<<ewGrid, blk256, 0, stream>>>(lt, w0 + lC, wrw, C, (int)MC);

    // a LoRA: aa = sigmoid(a0 + (xa@a1)@a2)
    launch_gemm(xa, a1 + (size_t)l * C * D, t1, M, D, C, 0, stream);
    launch_gemm(t1, a2 + (size_t)l * D * C, lt, M, C, D, 0, stream);
    ew_a_kernel<<<ewGrid, blk256, 0, stream>>>(lt, a0 + lC, aa, C, (int)MC);

    // v mixing
    if (l == 0) {
      copy_kernel<<<ewGrid, blk256, 0, stream>>>(vb, vfirst, (int)MC);
    } else {
      launch_gemm(xv, v1 + (size_t)l * C * D, t1, M, D, C, 0, stream);
      launch_gemm(t1, v2 + (size_t)l * D * C, lt, M, C, D, 0, stream);
      ew_vmix_kernel<<<ewGrid, blk256, 0, stream>>>(lt, v0 + lC, vfirst, vb, C, (int)MC);
    }

    // g LoRA: gg = sigmoid(xg@g1)@g2   (N=160 -> NBLK=32 path)
    launch_gemm(xg, g1 + (size_t)l * C * DG, t1, M, DG, C, 2, stream);
    launch_gemm(t1, g2 + (size_t)l * DG * C, gg, M, C, DG, 0, stream);

    // kk normalize + k rescale
    kk_prep_kernel<<<mhGrid, blk256, 0, stream>>>(kb, aa, k_k + lC, k_a + lC,
                                                  mkk, kka, M, H, N);
    // sequential state scan
    scan_kernel<<<B * H, 64, 0, stream>>>(rb, wrw, kb, vb, mkk, kka, yb, T, H, C);

    // groupnorm + rk*v + gate -> xr (reused as yg)
    post_y_kernel<<<mhGrid, blk256, 0, stream>>>(yb, rb, kb, vb, gg,
                                                 lnx_w + lC, lnx_b + lC, r_k + lC,
                                                 xr, M, H, N);
    // x += yg @ Wo
    launch_gemm(xr, Wo + (size_t)l * CC, x, M, C, C, 4, stream);

    // ---- channel mix ----
    ln_kernel<<<rowGrid, blk256, 0, stream>>>(x, ln2_w + lC, ln2_b + lC, xn, M, C);
    cmix_kernel<<<ewGrid, blk256, 0, stream>>>(xn, f_xk + lC, xk, T, C, (int)MC);
    launch_gemm(xk, f_key + (size_t)l * C * F4, hbuf, M, F4, C, 3, stream);  // relu^2
    launch_gemm(hbuf, f_val + (size_t)l * F4 * C, x, M, C, F4, 4, stream);   // x += h@f_val
  }

  // final LN + head
  ln_kernel<<<rowGrid, blk256, 0, stream>>>(x, lnout_w, lnout_b, xn, M, C);
  head_kernel<<<rowGrid, blk256, 0, stream>>>(xn, head_w, (float*)d_out, M, C);
}
